// VarianceAdaptor_45629732552877
// MI455X (gfx1250) — matrix operand order repaired and checked by reading, not compiled, not run on hardware
//
#include <hip/hip_runtime.h>
#include <hip/hip_bf16.h>

// ---------------------------------------------------------------------------
// Problem constants
// ---------------------------------------------------------------------------
#define BB  64
#define SS  512
#define DD  384
#define FF  256
#define MM  2048
#define NB  256          // NBINS
#define EPSLN 1e-5f

typedef __attribute__((ext_vector_type(16))) __bf16 v16bf;
typedef __attribute__((ext_vector_type(8)))  float  v8f;
typedef __attribute__((ext_vector_type(4)))  float  f32x4;

union FragAB { v16bf v; f32x4 q[2]; };

__device__ __forceinline__ unsigned short f2bf(float f) {
    union { float f; unsigned int u; } c; c.f = f;
    unsigned int u = c.u;
    u += 0x7FFFu + ((u >> 16) & 1u);          // round-to-nearest-even
    return (unsigned short)(u >> 16);
}
__device__ __forceinline__ float bf2f(unsigned short h) {
    union { unsigned int u; float f; } c; c.u = ((unsigned int)h) << 16;
    return c.f;
}

// ---------------------------------------------------------------------------
// Conv-as-GEMM via WMMA bf16.  A = halo-padded activations [B][S+2][Din]
// (im2col row s == contiguous 3*Din slice starting at padded row s),
// Bt = transposed weights [256][KD].  Out[b][s][f] = relu(A·B + bias).
//
// Block = 128 threads (4 waves); block tile = 32 rows x 256 cols.
// Each wave: 2 M-tiles x 4 N-tiles = 8 accumulators; B fragments are
// reused across both M-tiles -> 12 b128 loads per 8 WMMAs (1.5 loads/WMMA).
// ---------------------------------------------------------------------------
__global__ void gemm_conv_kernel(const __bf16* __restrict__ Apad,
                                 const __bf16* __restrict__ Bt,
                                 const float*  __restrict__ bias,
                                 float*        __restrict__ Out,
                                 int Din, int KD)
{
    const int lane = threadIdx.x & 31;
    const int wave = threadIdx.x >> 5;
    const int lr   = lane & 15;
    const int hl8  = (lane >> 4) * 8;    // unused (A and B loads both use hl16)
    const int hl16 = (lane >> 4) * 16;   // B-fragment half-lane K offset
    const int b    = blockIdx.y;
    const int m0   = blockIdx.x * 32;

    const __bf16* Arow0 = Apad + ((size_t)b * (SS + 2) + (m0 + lr)) * Din;
    const __bf16* Arow1 = Arow0 + (size_t)16 * Din;
    const int nb = wave * 64 + lr;
    const __bf16* B0 = Bt + (size_t)(nb +  0) * KD;
    const __bf16* B1 = Bt + (size_t)(nb + 16) * KD;
    const __bf16* B2 = Bt + (size_t)(nb + 32) * KD;
    const __bf16* B3 = Bt + (size_t)(nb + 48) * KD;

    v8f acc00 = (v8f)(0.0f), acc01 = (v8f)(0.0f), acc02 = (v8f)(0.0f), acc03 = (v8f)(0.0f);
    v8f acc10 = (v8f)(0.0f), acc11 = (v8f)(0.0f), acc12 = (v8f)(0.0f), acc13 = (v8f)(0.0f);

    for (int kk = 0; kk < KD; kk += 32) {
        FragAB a0, a1;
        a0.q[0] = *(const f32x4*)(Arow0 + kk + hl16);       // K kk+{0..7 | 16..23}
        a0.q[1] = *(const f32x4*)(Arow0 + kk + hl16 + 8);   // K kk+{8..15 | 24..31}
        a1.q[0] = *(const f32x4*)(Arow1 + kk + hl16);
        a1.q[1] = *(const f32x4*)(Arow1 + kk + hl16 + 8);

        FragAB f0, f1, f2, f3;
        f0.q[0] = *(const f32x4*)(B0 + kk + hl16);
        f0.q[1] = *(const f32x4*)(B0 + kk + hl16 + 8);
        f1.q[0] = *(const f32x4*)(B1 + kk + hl16);
        f1.q[1] = *(const f32x4*)(B1 + kk + hl16 + 8);
        f2.q[0] = *(const f32x4*)(B2 + kk + hl16);
        f2.q[1] = *(const f32x4*)(B2 + kk + hl16 + 8);
        f3.q[0] = *(const f32x4*)(B3 + kk + hl16);
        f3.q[1] = *(const f32x4*)(B3 + kk + hl16 + 8);

        acc00 = __builtin_amdgcn_wmma_f32_16x16x32_bf16(false, a0.v, false, f0.v, (short)0, acc00, false, false);
        acc10 = __builtin_amdgcn_wmma_f32_16x16x32_bf16(false, a1.v, false, f0.v, (short)0, acc10, false, false);
        acc01 = __builtin_amdgcn_wmma_f32_16x16x32_bf16(false, a0.v, false, f1.v, (short)0, acc01, false, false);
        acc11 = __builtin_amdgcn_wmma_f32_16x16x32_bf16(false, a1.v, false, f1.v, (short)0, acc11, false, false);
        acc02 = __builtin_amdgcn_wmma_f32_16x16x32_bf16(false, a0.v, false, f2.v, (short)0, acc02, false, false);
        acc12 = __builtin_amdgcn_wmma_f32_16x16x32_bf16(false, a1.v, false, f2.v, (short)0, acc12, false, false);
        acc03 = __builtin_amdgcn_wmma_f32_16x16x32_bf16(false, a0.v, false, f3.v, (short)0, acc03, false, false);
        acc13 = __builtin_amdgcn_wmma_f32_16x16x32_bf16(false, a1.v, false, f3.v, (short)0, acc13, false, false);
    }

    // Epilogue: C/D layout -> VGPR r: lanes 0-15 => (M=r, N=lane); 16-31 => (M=r+8, N=lane-16)
    const int rbase = (lane >> 4) * 8;
    v8f accs[2][4] = {{acc00, acc01, acc02, acc03}, {acc10, acc11, acc12, acc13}};
#pragma unroll
    for (int mt = 0; mt < 2; ++mt) {
        float* orow = Out + ((size_t)b * SS + m0 + mt * 16) * FF;
#pragma unroll
        for (int t = 0; t < 4; ++t) {
            const int col = wave * 64 + t * 16 + lr;
            const float bv = bias[col];
#pragma unroll
            for (int r = 0; r < 8; ++r) {
                float v = accs[mt][t][r] + bv;
                orow[(size_t)(rbase + r) * FF + col] = v > 0.f ? v : 0.f;
            }
        }
    }
}

// ---------------------------------------------------------------------------
// LayerNorm over F=256, one block per row; writes bf16 with optional halo pad.
// ---------------------------------------------------------------------------
__global__ void layernorm_kernel(const float* __restrict__ in,
                                 const float* __restrict__ g,
                                 const float* __restrict__ be,
                                 unsigned short* __restrict__ outb,
                                 int halo)
{
    __shared__ float red[FF];
    const int row = blockIdx.x;
    const int b = row / SS, s = row % SS;
    const int tid = threadIdx.x;

    float v = in[(size_t)row * FF + tid];
    red[tid] = v; __syncthreads();
#pragma unroll
    for (int o = FF / 2; o > 0; o >>= 1) { if (tid < o) red[tid] += red[tid + o]; __syncthreads(); }
    const float mean = red[0] * (1.0f / FF);
    __syncthreads();
    const float d = v - mean;
    red[tid] = d * d; __syncthreads();
#pragma unroll
    for (int o = FF / 2; o > 0; o >>= 1) { if (tid < o) red[tid] += red[tid + o]; __syncthreads(); }
    const float var = red[0] * (1.0f / FF);

    const float y = d * rsqrtf(var + EPSLN) * g[tid] + be[tid];
    outb[((size_t)b * (SS + 2 * halo) + s + halo) * FF + tid] = f2bf(y);
}

// Zero first + last padded row of each batch (halo rows for conv).
__global__ void zero_halo_kernel(unsigned short* __restrict__ buf, int rows, int Fd)
{
    size_t base = (size_t)blockIdx.x * rows * Fd;
    for (int i = threadIdx.x; i < Fd; i += blockDim.x) {
        buf[base + i] = 0;
        buf[base + (size_t)(rows - 1) * Fd + i] = 0;
    }
}

// f32 -> bf16 with halo offset (+1 row per batch)
__global__ void cvt_pad_kernel(const float* __restrict__ x,
                               unsigned short* __restrict__ xb, int Dd)
{
    const int row = blockIdx.x;
    const int b = row / SS, s = row % SS;
    const float* src = x + (size_t)row * Dd;
    unsigned short* dst = xb + ((size_t)b * (SS + 2) + s + 1) * Dd;
    for (int i = threadIdx.x; i < Dd; i += blockDim.x) dst[i] = f2bf(src[i]);
}

// w[KD][F] (f32) -> wt[F][KD] (bf16)
__global__ void wtrans_kernel(const float* __restrict__ w,
                              unsigned short* __restrict__ wt, int KD)
{
    size_t idx = (size_t)blockIdx.x * blockDim.x + threadIdx.x;
    if (idx >= (size_t)KD * FF) return;
    int kd = (int)(idx / FF), f = (int)(idx % FF);
    wt[(size_t)f * KD + kd] = f2bf(w[idx]);
}

// Final F->1 projection + mask; one block (256 thr) per row.
__global__ void linear_mask_kernel(const unsigned short* __restrict__ h,
                                   const float* __restrict__ wl,
                                   const float* __restrict__ bl,
                                   const unsigned char* __restrict__ mask,
                                   float* __restrict__ out)
{
    __shared__ float red[FF];
    const int row = blockIdx.x;
    const int tid = threadIdx.x;
    red[tid] = bf2f(h[(size_t)row * FF + tid]) * wl[tid];
    __syncthreads();
#pragma unroll
    for (int o = FF / 2; o > 0; o >>= 1) { if (tid < o) red[tid] += red[tid + o]; __syncthreads(); }
    if (tid == 0) out[row] = mask[row] ? 0.f : (red[0] + bl[0]);
}

__device__ __forceinline__ int bsearch_left(const float* bins, int n, float v)
{   // first i with bins[i] >= v
    int lo = 0, hi = n;
    while (lo < hi) { int mid = (lo + hi) >> 1; if (bins[mid] < v) lo = mid + 1; else hi = mid; }
    return lo;
}

// xmid = x + pitch_emb[idx]; also refresh bf16 halo-padded copy for energy conv
__global__ void pitch_add_kernel(const float* __restrict__ x,
                                 const float* __restrict__ target,
                                 const float* __restrict__ bins,
                                 const float* __restrict__ emb,
                                 float* __restrict__ xmid,
                                 unsigned short* __restrict__ xb2)
{
    const int row = blockIdx.x;
    const int b = row / SS, s = row % SS;
    const int idx = bsearch_left(bins, NB - 1, target[row]);
    const float* e   = emb + (size_t)idx * DD;
    const float* src = x + (size_t)row * DD;
    float* dm = xmid + (size_t)row * DD;
    unsigned short* db = xb2 + ((size_t)b * (SS + 2) + s + 1) * DD;
    for (int i = threadIdx.x; i < DD; i += blockDim.x) {
        float v = src[i] + e[i];
        dm[i] = v; db[i] = f2bf(v);
    }
}

// xmid += energy_emb[idx]  (in place; xmid fully rewritten earlier each call)
__global__ void energy_add_kernel(const float* __restrict__ target,
                                  const float* __restrict__ bins,
                                  const float* __restrict__ emb,
                                  float* __restrict__ xmid)
{
    const int row = blockIdx.x;
    const int idx = bsearch_left(bins, NB - 1, target[row]);
    const float* e = emb + (size_t)idx * DD;
    float* dm = xmid + (size_t)row * DD;
    for (int i = threadIdx.x; i < DD; i += blockDim.x) dm[i] += e[i];
}

// Inclusive scan of durations per batch (Hillis-Steele, 512 threads).
__global__ void cumsum_kernel(const int* __restrict__ dur,
                              int* __restrict__ cum,
                              float* __restrict__ mel_len)
{
    __shared__ int buf[SS];
    const int b = blockIdx.x, t = threadIdx.x;
    buf[t] = dur[b * SS + t];
    __syncthreads();
    for (int o = 1; o < SS; o <<= 1) {
        int v = (t >= o) ? buf[t - o] : 0;
        __syncthreads();
        buf[t] += v;
        __syncthreads();
    }
    cum[b * SS + t] = buf[t];
    if (t == SS - 1) mel_len[b] = (float)buf[t];
}

__global__ void durcopy_kernel(const int* __restrict__ dur, float* __restrict__ out)
{
    int i = blockIdx.x * blockDim.x + threadIdx.x;
    if (i < BB * SS) out[i] = (float)dur[i];
}

// Length regulation: per (b, frame) binary search + 384-float copy.
__global__ void gather_kernel(const float* __restrict__ xmid,
                              const int* __restrict__ cum,
                              float* __restrict__ out,
                              float* __restrict__ mel_mask)
{
    const int b = blockIdx.y, m = blockIdx.x;
    const int* c = cum + b * SS;
    int lo = 0, hi = SS;                    // searchsorted right: first i with c[i] > m
    while (lo < hi) { int mid = (lo + hi) >> 1; if (c[mid] <= m) lo = mid + 1; else hi = mid; }
    const bool valid = m < c[SS - 1];
    const int src = lo < SS - 1 ? lo : SS - 1;
    const float scale = valid ? 1.f : 0.f;
    const float* xs = xmid + ((size_t)b * SS + src) * DD;
    float* o = out + ((size_t)b * MM + m) * DD;
    for (int i = threadIdx.x; i < DD; i += blockDim.x) o[i] = xs[i] * scale;
    if (threadIdx.x == 0) mel_mask[b * MM + m] = valid ? 0.f : 1.f;
}

// ---------------------------------------------------------------------------
extern "C" void kernel_launch(void* const* d_in, const int* in_sizes, int n_in,
                              void* d_out, int out_size, void* d_ws, size_t ws_size,
                              hipStream_t stream)
{
    const float*         x        = (const float*)d_in[0];
    const unsigned char* src_mask = (const unsigned char*)d_in[1];
    const float*         pitch_t  = (const float*)d_in[2];
    const float*         energy_t = (const float*)d_in[3];
    const int*           dur_t    = (const int*)d_in[4];
    const float* dp[10]; const float* pp[10]; const float* ep[10];
    for (int i = 0; i < 10; ++i) {
        dp[i] = (const float*)d_in[6 + i];
        pp[i] = (const float*)d_in[16 + i];
        ep[i] = (const float*)d_in[26 + i];
    }
    const float* pitch_bins = (const float*)d_in[36];
    const float* energy_bins = (const float*)d_in[37];
    const float* pitch_emb  = (const float*)d_in[38];
    const float* energy_emb = (const float*)d_in[39];
    float* out = (float*)d_out;

    // d_out layout (reference return order, flattened as f32)
    const size_t OUT_OFF     = 0;
    const size_t PITCH_OFF   = OUT_OFF + (size_t)BB * MM * DD;
    const size_t ENERGY_OFF  = PITCH_OFF + (size_t)BB * SS;
    const size_t LOGDUR_OFF  = ENERGY_OFF + (size_t)BB * SS;
    const size_t DURT_OFF    = LOGDUR_OFF + (size_t)BB * SS;
    const size_t MELLEN_OFF  = DURT_OFF + (size_t)BB * SS;
    const size_t MELMASK_OFF = MELLEN_OFF + (size_t)BB;

    // Workspace carve
    char* ws = (char*)d_ws;
    size_t off = 0;
    auto carve = [&](size_t bytes) { void* p = ws + off; off += (bytes + 255) & ~(size_t)255; return p; };
    unsigned short* XB   = (unsigned short*)carve((size_t)BB * (SS + 2) * DD * 2); // bf16 x padded (reused as x2)
    float*          XMID = (float*)carve((size_t)BB * SS * DD * 4);                // x + embeddings (f32)
    float*          HF   = (float*)carve((size_t)BB * SS * FF * 4);                // gemm f32 staging (both convs)
    unsigned short* H1B  = (unsigned short*)carve((size_t)BB * (SS + 2) * FF * 2); // LN1 out, padded bf16
    unsigned short* H2B  = (unsigned short*)carve((size_t)BB * SS * FF * 2);       // LN2 out bf16
    unsigned short* WT1[3]; unsigned short* WT2[3];
    for (int i = 0; i < 3; ++i) {
        WT1[i] = (unsigned short*)carve((size_t)FF * 3 * DD * 2);
        WT2[i] = (unsigned short*)carve((size_t)FF * 3 * FF * 2);
    }
    int* CUM = (int*)carve((size_t)BB * SS * 4);

    const int KD2 = 3 * FF;
    dim3 ggrid(SS / 32, BB);

    auto run_predictor = [&](const unsigned short* xin, int Din,
                             const float* const* P, unsigned short* wt1,
                             unsigned short* wt2, float* pred_out) {
        int kd1 = 3 * Din;
        wtrans_kernel<<<(kd1 * FF + 255) / 256, 256, 0, stream>>>(P[0], wt1, kd1);
        wtrans_kernel<<<(KD2 * FF + 255) / 256, 256, 0, stream>>>(P[4], wt2, KD2);
        gemm_conv_kernel<<<ggrid, 128, 0, stream>>>((const __bf16*)xin, (const __bf16*)wt1, P[1], HF, Din, kd1);
        zero_halo_kernel<<<BB, 256, 0, stream>>>(H1B, SS + 2, FF);
        layernorm_kernel<<<BB * SS, FF, 0, stream>>>(HF, P[2], P[3], H1B, 1);
        gemm_conv_kernel<<<ggrid, 128, 0, stream>>>((const __bf16*)H1B, (const __bf16*)wt2, P[5], HF, FF, KD2);
        layernorm_kernel<<<BB * SS, FF, 0, stream>>>(HF, P[6], P[7], H2B, 0);
        linear_mask_kernel<<<BB * SS, FF, 0, stream>>>(H2B, P[8], P[9], src_mask, pred_out);
    };

    // bf16 halo-padded x
    cvt_pad_kernel<<<BB * SS, 128, 0, stream>>>(x, XB, DD);
    zero_halo_kernel<<<BB, 256, 0, stream>>>(XB, SS + 2, DD);

    // duration + pitch predictors on original x
    run_predictor(XB, DD, dp, WT1[0], WT2[0], out + LOGDUR_OFF);
    run_predictor(XB, DD, pp, WT1[1], WT2[1], out + PITCH_OFF);

    // x2 = x + pitch_emb[searchsorted]; XB reused as bf16 copy (halos still zero)
    pitch_add_kernel<<<BB * SS, 128, 0, stream>>>(x, pitch_t, pitch_bins, pitch_emb, XMID, XB);

    // energy predictor on x2
    run_predictor(XB, DD, ep, WT1[2], WT2[2], out + ENERGY_OFF);

    // x3 = x2 + energy_emb[searchsorted]  (in place)
    energy_add_kernel<<<BB * SS, 128, 0, stream>>>(energy_t, energy_bins, energy_emb, XMID);

    // duration path
    cumsum_kernel<<<BB, SS, 0, stream>>>(dur_t, CUM, out + MELLEN_OFF);
    durcopy_kernel<<<(BB * SS + 255) / 256, 256, 0, stream>>>(dur_t, out + DURT_OFF);
    gather_kernel<<<dim3(MM, BB), 128, 0, stream>>>(XMID, CUM, out + OUT_OFF, out + MELMASK_OFF);

    (void)in_sizes; (void)n_in; (void)out_size; (void)ws_size;
}
